// GraphConvolution_68547678044409
// MI455X (gfx1250) — matrix operand order repaired and checked by reading, not compiled, not run on hardware
//
#include <hip/hip_runtime.h>

// Problem sizes (fixed by the reference).
#define BB 16
#define SS 512
#define DD 512
#define OO 512
#define EE 8

typedef __attribute__((ext_vector_type(16))) _Float16 v16h;
typedef __attribute__((ext_vector_type(8)))  float    v8f;

// 32-byte fragment: 8 VGPRs of packed f16 (16 elements), loadable as 2x uint4.
union Frag {
    uint4    u[2];
    v16h     v;
    _Float16 h[16];
};

static __device__ __forceinline__ unsigned pack2h(float a, float b) {
    union { _Float16 h[2]; unsigned u; } p;
    p.h[0] = (_Float16)a;
    p.h[1] = (_Float16)b;
    return p.u;
}

// ---------------------------------------------------------------------------
// Kernel 1: x (f32) -> xh (f16), elementwise.
// ---------------------------------------------------------------------------
__global__ void cvt_x_kernel(const float* __restrict__ x,
                             _Float16* __restrict__ xh, int n) {
    int i = blockIdx.x * blockDim.x + threadIdx.x;
    int stride = gridDim.x * blockDim.x;
    for (; i < n; i += stride) xh[i] = (_Float16)x[i];
}

// ---------------------------------------------------------------------------
// Kernel 2: WcT[e,o,d] = (f16) Wc[e,d,o]  (transpose so B-fragments are
// K-contiguous per column; writes coalesced, reads served by L2).
// ---------------------------------------------------------------------------
__global__ void build_wct_kernel(const float* __restrict__ Wc,
                                 _Float16* __restrict__ wct) {
    int i = blockIdx.x * blockDim.x + threadIdx.x;   // over E*O*D
    if (i >= EE * OO * DD) return;
    int d = i % DD;
    int o = (i / DD) % OO;
    int e = i / (DD * OO);
    wct[i] = (_Float16)Wc[(e * DD + d) * OO + o];
}

// ---------------------------------------------------------------------------
// Kernel 3: gates[b,e,s] = sigmoid(x[b,s,:] . Wg[e,:] + bg[e])   (f32)
// 67 MFLOP total; x is L2-resident across the 8 edge types.
// ---------------------------------------------------------------------------
__global__ void gates_kernel(const float* __restrict__ x,
                             const float* __restrict__ Wg,
                             const float* __restrict__ bg,
                             float* __restrict__ gates) {
    int i = blockIdx.x * blockDim.x + threadIdx.x;   // over B*E*S
    if (i >= BB * EE * SS) return;
    int s = i % SS;
    int e = (i / SS) % EE;
    int b = i / (SS * EE);
    const float4* xr = (const float4*)(x + (size_t)(b * SS + s) * DD);
    const float4* wr = (const float4*)(Wg + e * DD);
    float z = bg[e];
#pragma unroll 4
    for (int q = 0; q < DD / 4; ++q) {
        float4 a = xr[q], w = wr[q];
        z += a.x * w.x + a.y * w.y + a.z * w.z + a.w * w.w;
    }
    gates[i] = 1.0f / (1.0f + __expf(-z));
}

// ---------------------------------------------------------------------------
// Kernel 4: hT[b,e,o,s] = (f16)( xh[b] @ WcT[e]^T + bc[e] )
// Block = (b, e, 32-row s tile), 8 waves; wave w owns o-chunk [64w, 64w+64).
// Per wave: 2(M) x 4(N) 16x16 f32 accumulators, K-loop of 32 (one WMMA depth).
// Transposed epilogue store is a single contiguous 16B store per lane.
// ---------------------------------------------------------------------------
__global__ __launch_bounds__(256) void h_gemm_kernel(
        const _Float16* __restrict__ xh, const _Float16* __restrict__ wct,
        const float* __restrict__ bc, _Float16* __restrict__ hT) {
    int blk  = blockIdx.x;          // B*E*(S/32) = 2048
    int st   = blk & 15;
    int e    = (blk >> 4) & 7;
    int b    = blk >> 7;
    int wave = threadIdx.x >> 5;
    int lane = threadIdx.x & 31;
    int col  = lane & 15;
    int half = lane >> 4;
    int s0 = st * 32;
    int o0 = wave * 64;

    v8f zero = {};
    v8f acc[2][4];
#pragma unroll
    for (int m = 0; m < 2; ++m)
#pragma unroll
        for (int n = 0; n < 4; ++n) acc[m][n] = zero;

    const _Float16* arow[2];
    arow[0] = xh + (size_t)(b * SS + s0 + col) * DD;
    arow[1] = arow[0] + (size_t)16 * DD;
    const _Float16* brow[4];
#pragma unroll
    for (int n = 0; n < 4; ++n)
        brow[n] = wct + (size_t)(e * OO + o0 + n * 16 + col) * DD;

    for (int k0 = 0; k0 < DD; k0 += 32) {
        Frag a[2];
#pragma unroll
        for (int m = 0; m < 2; ++m) {
            // A-frag layout (ISA 7.12.2): lanes 0-15: K=[k0..k0+7]+[k0+16..k0+23]
            //                             lanes 16-31: shift both runs by 8.
            a[m].u[0] = *(const uint4*)(arow[m] + k0 + half * 8);
            a[m].u[1] = *(const uint4*)(arow[m] + k0 + 16 + half * 8);
        }
#pragma unroll
        for (int n = 0; n < 4; ++n) {
            Frag bf;   // B-frag: lane n = column n; K runs same as A-frag
            bf.u[0] = *(const uint4*)(brow[n] + k0 + half * 8);
            bf.u[1] = *(const uint4*)(brow[n] + k0 + 16 + half * 8);
#pragma unroll
            for (int m = 0; m < 2; ++m)
                acc[m][n] = __builtin_amdgcn_wmma_f32_16x16x32_f16(
                    false, a[m].v, false, bf.v, (short)0, acc[m][n], false, false);
        }
    }

#pragma unroll
    for (int n = 0; n < 4; ++n) {
        int o = o0 + n * 16 + col;
        float bias = bc[e * OO + o];
#pragma unroll
        for (int m = 0; m < 2; ++m) {
            uint4 w;   // C-tile: VGPR r = M&7 -> consecutive r = consecutive s
            w.x = pack2h(acc[m][n][0] + bias, acc[m][n][1] + bias);
            w.y = pack2h(acc[m][n][2] + bias, acc[m][n][3] + bias);
            w.z = pack2h(acc[m][n][4] + bias, acc[m][n][5] + bias);
            w.w = pack2h(acc[m][n][6] + bias, acc[m][n][7] + bias);
            _Float16* dst = hT + ((size_t)((b * EE + e) * OO + o)) * SS
                               + s0 + m * 16 + half * 8;
            *(uint4*)dst = w;
        }
    }
}

// ---------------------------------------------------------------------------
// Kernel 5: out[b,s,o] = relu( Σ_e (adj[b,e] ⊙ gate-cols) @ h[b,e] )
// Effective GEMM with K = E*S = 4096. Block = (b, 16-row s tile); its 8 waves
// cover all of O, so each adj element hits HBM once and siblings hit WGP$/L2.
// Gate scaling + f32->f16 conversion fused into the A-fragment build.
// ---------------------------------------------------------------------------
__global__ __launch_bounds__(256) void out_gemm_kernel(
        const float* __restrict__ adj, const float* __restrict__ gates,
        const _Float16* __restrict__ hT, float* __restrict__ out) {
    int blk  = blockIdx.x;          // B*(S/16) = 512
    int st   = blk & 31;
    int b    = blk >> 5;
    int wave = threadIdx.x >> 5;
    int lane = threadIdx.x & 31;
    int col  = lane & 15;
    int half = lane >> 4;
    int s0 = st * 16;
    int o0 = wave * 64;

    v8f zero = {};
    v8f acc[4];
#pragma unroll
    for (int n = 0; n < 4; ++n) acc[n] = zero;

    for (int e = 0; e < EE; ++e) {
        const float* arow = adj + ((size_t)((b * EE + e) * SS + s0 + col)) * SS;
        const float* grow = gates + (size_t)(b * EE + e) * SS;
        const _Float16* hbase = hT + ((size_t)(b * EE + e)) * OO * SS;
        for (int t0 = 0; t0 < SS; t0 += 32) {
            // Stream the next adj chunk toward the caches (global_prefetch_b8).
            __builtin_prefetch(arow + t0 + 64, 0, 1);

            Frag a;
            int kk0 = t0 + half * 8;        // first 8-wide K run for this lane
            int kk1 = kk0 + 16;             // second run
            float4 A0 = *(const float4*)(arow + kk0);
            float4 A1 = *(const float4*)(arow + kk0 + 4);
            float4 G0 = *(const float4*)(grow + kk0);
            float4 G1 = *(const float4*)(grow + kk0 + 4);
            a.h[0] = (_Float16)(A0.x * G0.x);
            a.h[1] = (_Float16)(A0.y * G0.y);
            a.h[2] = (_Float16)(A0.z * G0.z);
            a.h[3] = (_Float16)(A0.w * G0.w);
            a.h[4] = (_Float16)(A1.x * G1.x);
            a.h[5] = (_Float16)(A1.y * G1.y);
            a.h[6] = (_Float16)(A1.z * G1.z);
            a.h[7] = (_Float16)(A1.w * G1.w);
            float4 A2 = *(const float4*)(arow + kk1);
            float4 A3 = *(const float4*)(arow + kk1 + 4);
            float4 G2 = *(const float4*)(grow + kk1);
            float4 G3 = *(const float4*)(grow + kk1 + 4);
            a.h[8]  = (_Float16)(A2.x * G2.x);
            a.h[9]  = (_Float16)(A2.y * G2.y);
            a.h[10] = (_Float16)(A2.z * G2.z);
            a.h[11] = (_Float16)(A2.w * G2.w);
            a.h[12] = (_Float16)(A3.x * G3.x);
            a.h[13] = (_Float16)(A3.y * G3.y);
            a.h[14] = (_Float16)(A3.z * G3.z);
            a.h[15] = (_Float16)(A3.w * G3.w);

#pragma unroll
            for (int n = 0; n < 4; ++n) {
                Frag bf;
                const _Float16* q = hbase
                    + (size_t)(o0 + n * 16 + col) * SS + t0 + half * 8;
                bf.u[0] = *(const uint4*)(q);
                bf.u[1] = *(const uint4*)(q + 16);
                acc[n] = __builtin_amdgcn_wmma_f32_16x16x32_f16(
                    false, a.v, false, bf.v, (short)0, acc[n], false, false);
            }
        }
    }

#pragma unroll
    for (int n = 0; n < 4; ++n) {
        int o = o0 + n * 16 + col;
#pragma unroll
        for (int r = 0; r < 8; ++r) {       // row M = half*8 + r
            float v = acc[n][r];
            v = v > 0.0f ? v : 0.0f;        // relu
            out[((size_t)(b * SS) + s0 + half * 8 + r) * OO + o] = v;
        }
    }
}

// ---------------------------------------------------------------------------
// Host launcher.
// Workspace layout (77 MiB):
//   [ 0,  8 MiB) xh   : x as f16                 (B*S*D*2  = 8 MiB)
//   [ 8, 12 MiB) wct  : Wc transposed, f16       (E*O*D*2  = 4 MiB)
//   [12, 13 MiB) gates: f32                      (B*E*S*4  = 256 KiB)
//   [13, 77 MiB) hT   : h transposed (b,e,o,s) f16 (B*E*O*S*2 = 64 MiB)
// ---------------------------------------------------------------------------
extern "C" void kernel_launch(void* const* d_in, const int* in_sizes, int n_in,
                              void* d_out, int out_size, void* d_ws, size_t ws_size,
                              hipStream_t stream) {
    const float* x   = (const float*)d_in[0];
    const float* adj = (const float*)d_in[1];
    const float* Wg  = (const float*)d_in[2];
    const float* bg  = (const float*)d_in[3];
    const float* Wc  = (const float*)d_in[4];
    const float* bc  = (const float*)d_in[5];
    float* out = (float*)d_out;

    char* ws = (char*)d_ws;
    _Float16* xh    = (_Float16*)(ws);
    _Float16* wct   = (_Float16*)(ws + (size_t)8  * 1024 * 1024);
    float*    gts   = (float*)   (ws + (size_t)12 * 1024 * 1024);
    _Float16* hT    = (_Float16*)(ws + (size_t)13 * 1024 * 1024);

    cvt_x_kernel<<<1024, 256, 0, stream>>>(x, xh, BB * SS * DD);
    build_wct_kernel<<<(EE * OO * DD) / 256, 256, 0, stream>>>(Wc, wct);
    gates_kernel<<<(BB * EE * SS) / 256, 256, 0, stream>>>(x, Wg, bg, gts);
    h_gemm_kernel<<<BB * EE * (SS / 32), 256, 0, stream>>>(xh, wct, bc, hT);
    out_gemm_kernel<<<BB * (SS / 16), 256, 0, stream>>>(adj, gts, hT, out);
}
